// ZINC_GIN_Inner_9534827397805
// MI455X (gfx1250) — matrix operand order repaired and checked by reading, not compiled, not run on hardware
//
#include <hip/hip_runtime.h>
#include <hip/hip_bf16.h>

typedef __attribute__((ext_vector_type(16))) _Float16 v16h;
typedef __attribute__((ext_vector_type(8)))  float    v8f;

#define N_NODES 200000
#define N_EDGES 600000
#define INF_    28
#define HID_    128
#define SSEG    20000

static_assert(N_NODES % 64 == 0, "node tiles exact");
static_assert(N_EDGES % 64 == 0, "edge tiles exact");

// ---------------- TDM weight loader (gfx1250 Tensor Data Mover) ----------------
#if defined(__AMDGCN__) && __has_builtin(__builtin_amdgcn_tensor_load_to_lds)
#define HAVE_TDM 1
#else
#define HAVE_TDM 0
#endif

#if HAVE_TDM
typedef __attribute__((ext_vector_type(4))) unsigned int tdm_u32x4;
typedef __attribute__((ext_vector_type(8))) int          tdm_i32x8;
typedef __attribute__((ext_vector_type(4))) int          tdm_i32x4;

// DMA a [KP][NP] f16 weight image global->LDS, hardware-padding each row by
// 8 halves (4 DWORDs) so the LDS leading dim becomes NP+8 (bank de-conflict).
template<int KP, int NP>
__device__ __forceinline__ void tdm_load_weights(const _Float16* __restrict__ gsrc,
                                                 _Float16* __restrict__ ldst)
{
  const unsigned long long ga = (unsigned long long)(uintptr_t)gsrc;
  const unsigned int la = (unsigned int)(uintptr_t)ldst; // flat LDS addr low 32 = LDS offset
  constexpr unsigned PAD_INT = (NP == 128) ? 5u : 3u;    // 64 / 16 DWORDs per row
  constexpr unsigned PAD_AMT = 3u;                       // 4 DWORDs = 8 halves of pad
  tdm_u32x4 g0;
  g0[0] = 1u;                                            // count=1, user descriptor
  g0[1] = la;                                            // lds_addr
  g0[2] = (unsigned)(ga & 0xFFFFFFFFull);                // global_addr[31:0]
  g0[3] = (unsigned)((ga >> 32) & 0x01FFFFFFull) | (2u << 30); // addr[56:32] | type=2
  tdm_i32x8 g1;
  g1[0] = (int)((1u << 16) | (1u << 20) | (PAD_INT << 22) | (PAD_AMT << 25));
          // data_size=2B, pad_enable, pad_interval, pad_amount
  g1[1] = (int)((unsigned)NP << 16);  // tensor_dim0[15:0]  @ bits 63:48
  g1[2] = (int)((unsigned)KP << 16);  // tensor_dim1[15:0]  @ bits 95:80
  g1[3] = (int)((unsigned)NP << 16);  // tile_dim0          @ bits 127:112
  g1[4] = (int)(unsigned)KP;          // tile_dim1          @ bits 143:128
  g1[5] = (int)(unsigned)NP;          // tensor_dim0_stride @ bits 191:160
  g1[6] = 0;
  g1[7] = 0;
  tdm_i32x4 z4 = {0, 0, 0, 0};
  tdm_i32x8 z8 = {0, 0, 0, 0, 0, 0, 0, 0};
  __builtin_amdgcn_tensor_load_to_lds(g0, g1, z4, z4, z8, 0);
}
#endif

// ---- WMMA fragment loaders (wave32, v_wmma_f32_16x16x32_f16 layouts) ----
__device__ __forceinline__ v16h load_a_frag(const _Float16* __restrict__ base, int ld, int lane) {
  const int m = lane & 15, hs = lane >> 4;
  const _Float16* p0 = base + m * ld + (hs << 4);
  const _Float16* p1 = p0 + 8;
  v16h a;
#pragma unroll
  for (int j = 0; j < 8; ++j) { a[j] = p0[j]; a[8 + j] = p1[j]; }
  return a;
}

__device__ __forceinline__ v16h load_b_frag(const _Float16* __restrict__ base, int ld, int lane) {
  const int nl = lane & 15, hs = lane >> 4;
  const _Float16* p = base + (hs << 4) * ld + nl;
  v16h b;
#pragma unroll
  for (int j = 0; j < 16; ++j) b[j] = p[j * ld];
  return b;
}

// ---- pre-convert weights: f32 [K][N] -> zero-padded f16 [KP][NP] image ----
__global__ void convert_w_kernel(const float* __restrict__ W, int K, int N,
                                 int KP, int NP, _Float16* __restrict__ out)
{
  int i = blockIdx.x * blockDim.x + threadIdx.x;
  if (i < KP * NP) {
    int k = i / NP, n = i % NP;
    out[i] = (_Float16)((k < K && n < N) ? W[k * N + n] : 0.f);
  }
}

// ======================================================================
// Fused bond-encoder MLP + message + scatter-add.
//   e  = relu(edge_attr @ W1 + b1) @ W2 + b2        (E x NV)
//   m  = relu(x[src] + e) * w ; atomicAdd aggr[dst] (L2-resident scatter)
// 64 edges per block, 256 threads = 8 wave32s. W2h pre-padded f16 [KP][KP].
// ======================================================================
template<int KP, int NV>
__global__ __launch_bounds__(256)
void edge_message_kernel(const float* __restrict__ x,
                         const int*   __restrict__ ei,
                         const float* __restrict__ ea,
                         const float* __restrict__ ew,
                         const float* __restrict__ W1,
                         const float* __restrict__ b1,
                         const _Float16* __restrict__ W2h,
                         const float* __restrict__ b2,
                         float* __restrict__ aggr)
{
  constexpr int LDA = KP + 8;
  __shared__ _Float16 sA[64 * LDA];    // relu(ea@W1+b1), f16, K-padded
  __shared__ _Float16 sW[KP * LDA];    // W2 f16 (TDM-padded rows)
  __shared__ float sW1[4 * NV];
  __shared__ float sB1[KP];
  __shared__ float sB2[KP];
  __shared__ float sEA[64 * 4];
  __shared__ int   sSrc[64];
  __shared__ int   sDst[64];
  __shared__ float sWgt[64];

  const int tid = threadIdx.x;
  const int e0  = blockIdx.x * 64;

#if HAVE_TDM
  if (tid == 0) tdm_load_weights<KP, KP>(W2h, sW);
#else
  for (int i = tid; i < KP * KP; i += 256) {
    int k = i / KP, n = i % KP;
    sW[k * LDA + n] = W2h[i];
  }
#endif

  for (int i = tid; i < 4 * NV; i += 256) sW1[i] = W1[i];
  for (int i = tid; i < KP; i += 256) {
    sB1[i] = (i < NV) ? b1[i] : 0.f;
    sB2[i] = (i < NV) ? b2[i] : 0.f;
  }
  for (int i = tid; i < 64; i += 256) {
    int e = e0 + i;
    int s = ei[e];
    sSrc[i] = s;
    sDst[i] = ei[N_EDGES + e];
    sWgt[i] = ew[e];
    __builtin_prefetch(&x[(size_t)s * NV], 0, 3);   // warm the gather rows
  }
  for (int i = tid; i < 64 * 4; i += 256) {
    int e = e0 + (i >> 2);
    sEA[i] = ea[(size_t)e * 4 + (i & 3)];
  }
#if HAVE_TDM
  __builtin_amdgcn_s_wait_tensorcnt((short)0);       // sW resident before barrier
#endif
  __syncthreads();

  // tiny K=4 first layer of bond encoder (VALU), relu, convert f16 -> LDS A
  for (int i = tid; i < 64 * KP; i += 256) {
    int r = i / KP, c = i % KP;
    float v = 0.f;
    if (c < NV) {
      float acc = sB1[c];
#pragma unroll
      for (int k = 0; k < 4; ++k) acc = fmaf(sEA[r * 4 + k], sW1[k * NV + c], acc);
      v = fmaxf(acc, 0.f);
    }
    sA[r * LDA + c] = (_Float16)v;
  }
  __syncthreads();

  const int wave = tid >> 5, lane = tid & 31;
  const int nl = lane & 15, hs = lane >> 4;
  constexpr int NT = KP / 16;
  constexpr int TRIPS = (4 * NT) / 8;   // 4 (KP=128) or 1 (KP=32)
#pragma unroll
  for (int tt = 0; tt < TRIPS; ++tt) {
    const int t = wave + tt * 8;
    const int rs = t / NT, nt = t % NT;
    const int row0 = rs * 16, n0 = nt * 16;
    v8f acc;
    {
      float bb = sB2[n0 + nl];
#pragma unroll
      for (int r = 0; r < 8; ++r) acc[r] = bb;
    }
#pragma unroll
    for (int kc = 0; kc < KP; kc += 32) {
      v16h a = load_a_frag(sA + row0 * LDA + kc, LDA, lane);
      v16h b = load_b_frag(sW + kc * LDA + n0, LDA, lane);
      acc = __builtin_amdgcn_wmma_f32_16x16x32_f16(false, a, false, b, (short)0, acc,
                                                   false, false);
    }
    // message + scatter straight out of the accumulators (never spill e)
    const int col = n0 + nl;
    if (NV == 128 || col < NV) {
#pragma unroll
      for (int r = 0; r < 8; ++r) {
        const int lr = row0 + r + (hs << 3);   // C layout: M = r + 8*hs
        const int s = sSrc[lr], d = sDst[lr];
        float m = fmaxf(x[(size_t)s * NV + col] + acc[r], 0.f) * sWgt[lr];
        atomicAdd(&aggr[(size_t)d * NV + col], m);
      }
    }
  }
}

// ======================================================================
// Node GEMM: out = maybe_relu( ((1+eps)*A + A2) @ W + bias ), N fixed 128.
// Optional fused BN statistics. Wh pre-padded f16 [KP][128].
// 64 rows per block, 256 threads = 8 wave32s, 4 output tiles per wave.
// ======================================================================
template<int KP>
__global__ __launch_bounds__(256)
void gemm_node_kernel(const float* __restrict__ A, int a_cols,
                      const float* __restrict__ A2,
                      const float* __restrict__ eps_p, int eps_idx,
                      const _Float16* __restrict__ Wh,
                      const float* __restrict__ bias,
                      float* __restrict__ out,
                      int relu_out,
                      float* __restrict__ bnsum, float* __restrict__ bnsumsq)
{
  constexpr int LDA = KP + 8;
  constexpr int LDW = HID_ + 8;
  __shared__ _Float16 sA[64 * LDA];
  __shared__ _Float16 sW[KP * LDW];
  __shared__ float sBias[HID_];
  __shared__ float sSum[HID_];
  __shared__ float sSq[HID_];

  const int tid = threadIdx.x;
  const int r0  = blockIdx.x * 64;
  const float scale = eps_p ? (1.f + eps_p[eps_idx]) : 1.f;

#if HAVE_TDM
  if (tid == 0) tdm_load_weights<KP, HID_>(Wh, sW);
#else
  for (int i = tid; i < KP * HID_; i += 256) {
    int k = i / HID_, n = i % HID_;
    sW[k * LDW + n] = Wh[i];
  }
#endif

  for (int i = tid; i < HID_; i += 256) { sBias[i] = bias[i]; sSum[i] = 0.f; sSq[i] = 0.f; }
  for (int i = tid; i < 64 * KP; i += 256) {
    int r = i / KP, c = i % KP;
    float v = 0.f;
    if (c < a_cols) {
      size_t idx = (size_t)(r0 + r) * a_cols + c;
      v = scale * A[idx];
      if (A2) v += A2[idx];
    }
    sA[r * LDA + c] = (_Float16)v;
  }
#if HAVE_TDM
  __builtin_amdgcn_s_wait_tensorcnt((short)0);
#endif
  __syncthreads();

  const int wave = tid >> 5, lane = tid & 31;
  const int nl = lane & 15, hs = lane >> 4;
#pragma unroll
  for (int tt = 0; tt < 4; ++tt) {
    const int t = wave + tt * 8;
    const int rs = t >> 3, nt = t & 7;
    const int row0 = rs * 16, n0 = nt * 16;
    v8f acc;
    {
      float bb = sBias[n0 + nl];
#pragma unroll
      for (int r = 0; r < 8; ++r) acc[r] = bb;
    }
#pragma unroll
    for (int kc = 0; kc < KP; kc += 32) {
      v16h a = load_a_frag(sA + row0 * LDA + kc, LDA, lane);
      v16h b = load_b_frag(sW + kc * LDW + n0, LDW, lane);
      acc = __builtin_amdgcn_wmma_f32_16x16x32_f16(false, a, false, b, (short)0, acc,
                                                   false, false);
    }
    const int col = n0 + nl;
    float csum = 0.f, csq = 0.f;
#pragma unroll
    for (int r = 0; r < 8; ++r) {
      const int row = r0 + row0 + r + (hs << 3);
      float v = acc[r];
      if (relu_out) v = fmaxf(v, 0.f);
      out[(size_t)row * HID_ + col] = v;
      csum += v; csq += v * v;
    }
    if (bnsum) {               // fused BN statistics (LDS f32 atomics)
      atomicAdd(&sSum[col], csum);
      atomicAdd(&sSq[col],  csq);
    }
  }
  if (bnsum) {
    __syncthreads();
    if (tid < HID_)            atomicAdd(&bnsum[tid], sSum[tid]);
    else if (tid < 2 * HID_)   atomicAdd(&bnsumsq[tid - HID_], sSq[tid - HID_]);
  }
}

// ---- BN finalize: mean/var -> scale/shift (one tiny block) ----
__global__ void bn_finalize_kernel(const float* __restrict__ sum,
                                   const float* __restrict__ sq,
                                   const float* __restrict__ g,
                                   const float* __restrict__ beta,
                                   float invn,
                                   float* __restrict__ scl,
                                   float* __restrict__ shf)
{
  int c = threadIdx.x;
  float mean = sum[c] * invn;
  float var  = fmaxf(sq[c] * invn - mean * mean, 0.f);
  float s    = g[c] * rsqrtf(var + 1e-5f);
  scl[c] = s;
  shf[c] = beta[c] - mean * s;
}

// ---- BN apply + ReLU (+ residual) ----
__global__ void bn_apply_kernel(const float* __restrict__ h2,
                                const float* __restrict__ scl,
                                const float* __restrict__ shf,
                                const float* __restrict__ resid,
                                float* __restrict__ out, int total)
{
  int i = blockIdx.x * blockDim.x + threadIdx.x;
  if (i < total) {
    int c = i & (HID_ - 1);
    float v = fmaxf(fmaf(h2[i], scl[c], shf[c]), 0.f);
    if (resid) v += resid[i];
    out[i] = v;
  }
}

// ---- masked global_add_pool over subgraphs (scatter into [S,128]) ----
__global__ void pool_kernel(const float* __restrict__ h,
                            const float* __restrict__ mask,
                            const int*   __restrict__ seg,
                            float* __restrict__ out, int total)
{
  int i = blockIdx.x * blockDim.x + threadIdx.x;
  if (i < total) {
    int n = i >> 7, c = i & 127;
    float m = mask[n];
    if (m != 0.f) atomicAdd(&out[(size_t)seg[n] * HID_ + c], h[i] * m);
  }
}

extern "C" void kernel_launch(void* const* d_in, const int* in_sizes, int n_in,
                              void* d_out, int out_size, void* d_ws, size_t ws_size,
                              hipStream_t stream)
{
  const float* x     = (const float*)d_in[0];
  const int*   ei    = (const int*)  d_in[1];
  const float* ea    = (const float*)d_in[2];
  const float* ew    = (const float*)d_in[3];
  const float* msk   = (const float*)d_in[4];
  const int*   seg   = (const int*)  d_in[5];
  const float* be1W1 = (const float*)d_in[6];
  const float* be1b1 = (const float*)d_in[7];
  const float* be1W2 = (const float*)d_in[8];
  const float* be1b2 = (const float*)d_in[9];
  const float* m1W1  = (const float*)d_in[10];
  const float* m1b1  = (const float*)d_in[11];
  const float* m1W2  = (const float*)d_in[12];
  const float* m1b2  = (const float*)d_in[13];
  const float* bn1g  = (const float*)d_in[14];
  const float* bn1b  = (const float*)d_in[15];
  const float* eps1  = (const float*)d_in[16];
  const float* beW1  = (const float*)d_in[17];
  const float* beb1  = (const float*)d_in[18];
  const float* beW2  = (const float*)d_in[19];
  const float* beb2  = (const float*)d_in[20];
  const float* mW1   = (const float*)d_in[21];
  const float* mb1   = (const float*)d_in[22];
  const float* mW2   = (const float*)d_in[23];
  const float* mb2   = (const float*)d_in[24];
  const float* bng   = (const float*)d_in[25];
  const float* bnb   = (const float*)d_in[26];
  const float* epsL  = (const float*)d_in[27];

  const size_t NH = (size_t)N_NODES * HID_;
  float* S0    = (float*)d_ws;       // activations ping
  float* S1    = S0 + NH;            // activations pong / layer-1 scratch
  float* S2    = S1 + NH;            // aggr / g1 / h2 scratch (hidden layers)
  float* bnsum = S2 + NH;
  float* bnsq  = bnsum + HID_;
  float* bnscl = bnsq + HID_;
  float* bnshf = bnscl + HID_;

  // pre-converted, zero-padded f16 weight images for WMMA / TDM
  _Float16* wf      = (_Float16*)(bnshf + HID_);
  _Float16* be1W2p  = wf;                        // [32][32]
  _Float16* m1W1p   = be1W2p + 32 * 32;          // [32][128]
  _Float16* m1W2p   = m1W1p  + 32 * 128;         // [128][128]
  _Float16* hidW    = m1W2p  + 128 * 128;        // 9 x [128][128]

  const dim3 blk(256);
  const int EB = N_EDGES / 64;            // 9375 (exact)
  const int NB = N_NODES / 64;            // 3125 (exact)
  const int EL = (N_NODES * HID_) / 256;  // 100000 (exact)
  const float invN = 1.f / (float)N_NODES;

  // ---------------- weight pre-conversion (f32 -> padded f16) ----------------
  convert_w_kernel<<<(32 * 32 + 255) / 256, blk, 0, stream>>>(be1W2, 28, 28, 32, 32, be1W2p);
  convert_w_kernel<<<(32 * 128 + 255) / 256, blk, 0, stream>>>(m1W1, 28, 128, 32, 128, m1W1p);
  convert_w_kernel<<<64, blk, 0, stream>>>(m1W2, 128, 128, 128, 128, m1W2p);
  for (int i = 0; i < 3; ++i) {
    convert_w_kernel<<<64, blk, 0, stream>>>(beW2 + (size_t)i * 16384, 128, 128, 128, 128,
                                             hidW + (size_t)(3 * i + 0) * 16384);
    convert_w_kernel<<<64, blk, 0, stream>>>(mW1 + (size_t)i * 16384, 128, 128, 128, 128,
                                             hidW + (size_t)(3 * i + 1) * 16384);
    convert_w_kernel<<<64, blk, 0, stream>>>(mW2 + (size_t)i * 16384, 128, 128, 128, 128,
                                             hidW + (size_t)(3 * i + 2) * 16384);
  }

  // ---------------- Layer 1 (28 -> 128, no residual) ----------------
  (void)hipMemsetAsync(S2, 0, (size_t)N_NODES * INF_ * sizeof(float), stream);
  (void)hipMemsetAsync(bnsum, 0, 2 * HID_ * sizeof(float), stream);
  edge_message_kernel<32, 28><<<EB, blk, 0, stream>>>(x, ei, ea, ew,
      be1W1, be1b1, be1W2p, be1b2, S2);
  gemm_node_kernel<32><<<NB, blk, 0, stream>>>(x, INF_, S2, eps1, 0,
      m1W1p, m1b1, S1, 1, nullptr, nullptr);
  gemm_node_kernel<128><<<NB, blk, 0, stream>>>(S1, HID_, nullptr, nullptr, 0,
      m1W2p, m1b2, S1, 0, bnsum, bnsq);
  bn_finalize_kernel<<<1, HID_, 0, stream>>>(bnsum, bnsq, bn1g, bn1b, invN, bnscl, bnshf);
  bn_apply_kernel<<<EL, blk, 0, stream>>>(S1, bnscl, bnshf, nullptr, S0, N_NODES * HID_);

  // ---------------- Layers 2..4 (128 -> 128, residual) ----------------
  float* P = S0;
  float* Q = S1;
  for (int i = 0; i < 3; ++i) {
    (void)hipMemsetAsync(S2, 0, NH * sizeof(float), stream);
    (void)hipMemsetAsync(bnsum, 0, 2 * HID_ * sizeof(float), stream);
    edge_message_kernel<128, 128><<<EB, blk, 0, stream>>>(P, ei, ea, ew,
        beW1 + (size_t)i * 4 * HID_, beb1 + (size_t)i * HID_,
        hidW + (size_t)(3 * i + 0) * 16384, beb2 + (size_t)i * HID_, S2);
    // g1 overwrites aggr in place (block-exclusive 64-row tiles, loads precede stores)
    gemm_node_kernel<128><<<NB, blk, 0, stream>>>(P, HID_, S2, epsL, i,
        hidW + (size_t)(3 * i + 1) * 16384, mb1 + (size_t)i * HID_, S2, 1,
        nullptr, nullptr);
    gemm_node_kernel<128><<<NB, blk, 0, stream>>>(S2, HID_, nullptr, nullptr, 0,
        hidW + (size_t)(3 * i + 2) * 16384, mb2 + (size_t)i * HID_, S2, 0,
        bnsum, bnsq);
    bn_finalize_kernel<<<1, HID_, 0, stream>>>(bnsum, bnsq,
        bng + (size_t)i * HID_, bnb + (size_t)i * HID_, invN, bnscl, bnshf);
    bn_apply_kernel<<<EL, blk, 0, stream>>>(S2, bnscl, bnshf, P, Q, N_NODES * HID_);
    float* tmp = P; P = Q; Q = tmp;
  }

  // ---------------- masked pool into d_out [S,128] ----------------
  (void)hipMemsetAsync(d_out, 0, (size_t)SSEG * HID_ * sizeof(float), stream);
  pool_kernel<<<EL, blk, 0, stream>>>(P, msk, seg, (float*)d_out, N_NODES * HID_);
}
